// TorchESN_953482740260
// MI455X (gfx1250) — matrix operand order repaired and checked by reading, not compiled, not run on hardware
//
#include <hip/hip_runtime.h>
#include <hip/hip_bf16.h>
#include <math.h>
#include <stdint.h>

typedef __bf16 bf16_t;
typedef __attribute__((ext_vector_type(16))) __bf16 v16bf;
typedef __attribute__((ext_vector_type(8)))  __bf16 v8bf;
typedef __attribute__((ext_vector_type(8)))  float  v8f;

#define T_DIM 4096
#define I_DIM 1024
#define H_DIM 4096
#define EXT_LD (1 + I_DIM + H_DIM)   // 5121, row stride of W_out

static __device__ __forceinline__ v16bf zero_v16bf() {
    v16bf z;
#pragma unroll
    for (int i = 0; i < 16; ++i) z[i] = (__bf16)0.0f;
    return z;
}

// Async global->LDS copy of 16B per lane (512B per wave-instruction),
// tracked by ASYNCcnt. lds = 32-bit LDS byte offset, g = 64-bit global addr.
#define ASYNC_G2L_B128(lds, g)                                             \
    asm volatile("global_load_async_to_lds_b128 %0, %1, off"               \
                 :: "v"(lds), "v"(g) : "memory")

// ---------------------------------------------------------------------------
// Init: h0 from `state` (bf16), zero second h buffer, extract readout bias
// (column 0 of W_out), reset grid-barrier counter.
// ---------------------------------------------------------------------------
__global__ void esn_init(const float* __restrict__ state,
                         const float* __restrict__ W_out,
                         bf16_t* __restrict__ hbuf,
                         float* __restrict__ bias_out,
                         unsigned* __restrict__ bar) {
    int idx = blockIdx.x * 256 + threadIdx.x;
    if (idx < H_DIM) {
        hbuf[idx]         = (bf16_t)state[idx];
        hbuf[H_DIM + idx] = (bf16_t)0.0f;
    }
    if (idx < I_DIM) bias_out[idx] = W_out[(size_t)idx * EXT_LD];
    if (idx == 0) *bar = 0u;
}

// ---------------------------------------------------------------------------
// Elementwise fp32 -> bf16 conversion (inputs X).
// ---------------------------------------------------------------------------
__global__ void conv_bf16(const float* __restrict__ src,
                          bf16_t* __restrict__ dst, long n) {
    long idx = (long)blockIdx.x * 256 + threadIdx.x;
    if (idx < n) dst[idx] = (bf16_t)src[idx];
}

// ---------------------------------------------------------------------------
// Pack a weight slice W[n][col_off + k] (row-major, row stride ldw) into the
// WMMA bf16 B-fragment order used here (k contiguous per lane half) for B[k][n] = W[n][k]:
//   tile (nt, kt) holds N=nt*16..+15, K=kt*32..+31, 512 bf16 per tile.
//   lane l (0..31): n = nt*16 + (l&15); k-range = kt*32 + (l>>4)*16 + (0..15)
// so each lane's 16 values are one contiguous 32-byte load at tile + l*16.
// ---------------------------------------------------------------------------
__global__ void pack_b(const float* __restrict__ W, int ldw, int col_off,
                       bf16_t* __restrict__ Bp, int KT /* K/32 */, long total) {
    long idx = (long)blockIdx.x * 256 + threadIdx.x;
    if (idx >= total) return;
    long tile   = idx >> 9;             // 512 elements per tile
    int  within = (int)(idx & 511);
    int  lane   = within >> 4;
    int  i      = within & 15;
    int  kt     = (int)(tile % KT);
    int  nt     = (int)(tile / KT);
    int  n = nt * 16 + (lane & 15);
    int  k = kt * 32 + (lane >> 4) * 16 + i;
    Bp[idx] = (bf16_t)W[(size_t)n * ldw + col_off + k];
}

// ---------------------------------------------------------------------------
// Tiled GEMM: C[M,N] (+)= A[M,K](bf16, row-major) * Bp (packed) [+ bias[n]].
// One wave owns a 32(M) x 16(N) block: two v_wmma_f32_16x16x32_bf16 per
// K-step sharing one B fragment. All 8 waves share one A tile, so the
// 32x32 A tile is staged in LDS via async global->LDS copies (wave 0,
// double-buffered, ASYNCcnt handoff) and read back with ds_load_b128.
// A-fragment layout (matches packed B): lane l: M = l&15,
// K = (l>>4)*16 + {0..15}.
// ---------------------------------------------------------------------------
__global__ __launch_bounds__(256)
void gemm_wmma_bf16(const bf16_t* __restrict__ A, int lda,
                    const bf16_t* __restrict__ Bp,
                    float* __restrict__ C, int ldc,
                    int K, const float* __restrict__ bias, int accum) {
    __shared__ bf16_t atile[2][32 * 32];          // double-buffered, 2x2KB
    const int lane  = threadIdx.x & 31;
    const int wave  = threadIdx.x >> 5;
    const int ntile = blockIdx.x * 8 + wave;
    const int m0    = blockIdx.y * 32;
    const int KT    = K >> 5;

    // async-copy addressing: one instr covers 8 rows (4 lanes x 16B per row)
    const unsigned long long gbase =
        (unsigned long long)(uintptr_t)A
        + (unsigned long long)(m0 + (lane >> 2)) * ((unsigned long long)lda * 2)
        + (unsigned)((lane & 3) * 16);
    const unsigned long long rowstep = (unsigned long long)lda * 16;  // 8 rows
    const unsigned lds0 = (unsigned)(uintptr_t)&atile[0][0] + lane * 16;

    if (wave == 0) {   // prologue: stage kt=0 into buffer 0
#pragma unroll
        for (int c = 0; c < 4; ++c)
            ASYNC_G2L_B128(lds0 + c * 512, gbase + (unsigned long long)c * rowstep);
    }

    const bf16_t* btile = Bp + (size_t)ntile * KT * 512 + lane * 16;
    const int r0    = lane & 15;
    const int khalf = (lane >> 4) * 16;

    v8f acc0 = {};
    v8f acc1 = {};
    for (int kt = 0; kt < KT; ++kt) {
        if (wave == 0) {
            if (kt + 1 < KT) {   // stage next tile, then wait for current
#pragma unroll
                for (int c = 0; c < 4; ++c)
                    ASYNC_G2L_B128(lds0 + (((kt + 1) & 1) * 2048) + c * 512,
                                   gbase + (unsigned long long)(kt + 1) * 64
                                         + (unsigned long long)c * rowstep);
                asm volatile("s_wait_asynccnt 0x4" ::: "memory");
            } else {
                asm volatile("s_wait_asynccnt 0x0" ::: "memory");
            }
        }
        __syncthreads();                           // tile kt visible to all

        const bf16_t* tb = &atile[kt & 1][0];
        union { v16bf v; v8bf h[2]; } a0, a1;
        a0.h[0] = *(const v8bf*)(tb + r0 * 32 + khalf);
        a0.h[1] = *(const v8bf*)(tb + r0 * 32 + 8 + khalf);
        a1.h[0] = *(const v8bf*)(tb + (16 + r0) * 32 + khalf);
        a1.h[1] = *(const v8bf*)(tb + (16 + r0) * 32 + 8 + khalf);
        v16bf b = *(const v16bf*)(btile + (size_t)kt * 512);
        acc0 = __builtin_amdgcn_wmma_f32_16x16x32_bf16(false, a0.v, false, b,
                                                       (short)0, acc0, false, false);
        acc1 = __builtin_amdgcn_wmma_f32_16x16x32_bf16(false, a1.v, false, b,
                                                       (short)0, acc1, false, false);
        __syncthreads();                  // reads done before buffer reuse
    }
    // C/D layout: VGPR r, lane l -> M = r + 8*(l>>4), N = l&15
    const int col   = ntile * 16 + (lane & 15);
    const int rbase = (lane >> 4) * 8;
    const float bv  = bias ? bias[col] : 0.0f;
    float* c0 = C + (size_t)(m0 + rbase) * ldc + col;
    float* c1 = C + (size_t)(m0 + 16 + rbase) * ldc + col;
#pragma unroll
    for (int r = 0; r < 8; ++r) {
        float v0 = acc0[r] + bv;
        float v1 = acc1[r] + bv;
        if (accum) { v0 += c0[(size_t)r * ldc]; v1 += c1[(size_t)r * ldc]; }
        c0[(size_t)r * ldc] = v0;
        c1[(size_t)r * ldc] = v1;
    }
}

// ---------------------------------------------------------------------------
// Persistent recurrence: h_t = tanh(P[t] + h_{t-1} * W_hh^T), 4096 steps.
// 32 blocks x 8 waves = 256 waves; wave -> one 16-column N tile of h.
// Per step, wave 0 async-copies the 8KB h vector into LDS once per block
// (vs 64KB/block of redundant per-wave global reads), then all waves run
// the GEMV via WMMA with h in row 0 of the A fragment. W_hh streams
// straight from L2 (wave-private, zero reuse -> no LDS staging). Grid-wide
// sense-free barrier (monotonic counter, release add / acquire spin with
// s_sleep) separates steps.
// ---------------------------------------------------------------------------
__global__ __launch_bounds__(256)
void esn_recurrence(const bf16_t* __restrict__ WhhP,
                    const float* __restrict__ P,
                    float* __restrict__ states,
                    bf16_t* __restrict__ Sbf,
                    bf16_t* __restrict__ hbuf,
                    unsigned* __restrict__ bar) {
    __shared__ bf16_t hlds[H_DIM];                      // 8 KB
    const int lane  = threadIdx.x & 31;
    const int wave  = threadIdx.x >> 5;
    const int ntile = blockIdx.x * 8 + wave;            // 0..255
    const int n     = ntile * 16 + (lane & 15);
    const int KT    = H_DIM >> 5;                       // 128
    const unsigned nb = gridDim.x;
    const bf16_t* btile = WhhP + (size_t)ntile * KT * 512 + lane * 16;
    const int  khalf = (lane >> 4) * 16;
    const bool row0  = ((lane & 15) == 0);
    const unsigned hlds0 = (unsigned)(uintptr_t)&hlds[0] + lane * 16;

    for (int t = 0; t < T_DIM; ++t) {
        // stage h_{t-1} (global, updated before last grid barrier) into LDS
        if (wave == 0) {
            const unsigned long long hg =
                (unsigned long long)(uintptr_t)(hbuf + (size_t)(t & 1) * H_DIM)
                + lane * 16;
#pragma unroll
            for (int c = 0; c < 16; ++c)               // 16 x 512B = 8KB
                ASYNC_G2L_B128(hlds0 + c * 512, hg + (unsigned long long)(c * 512));
            asm volatile("s_wait_asynccnt 0x0" ::: "memory");
        }
        __syncthreads();

        v8f acc = {};
        for (int kt = 0; kt < KT; ++kt) {
            union { v16bf v; v8bf h[2]; } a;
            if (row0) {   // lanes 0 / 16 carry row M=0 of the A fragment
                a.h[0] = *(const v8bf*)(hlds + kt * 32 + khalf);
                a.h[1] = *(const v8bf*)(hlds + kt * 32 + 8 + khalf);
            } else {
                a.v = zero_v16bf();
            }
            v16bf b = *(const v16bf*)(btile + (size_t)kt * 512);
            if (kt + 2 < KT)  // stream next W_hh tiles
                __builtin_prefetch((const void*)(btile + (size_t)(kt + 2) * 512), 0, 1);
            acc = __builtin_amdgcn_wmma_f32_16x16x32_bf16(false, a.v, false, b,
                                                          (short)0, acc, false, false);
        }
        // D row M=0 sits in VGPR0 of lanes 0..15 (N = lane)
        if (lane < 16) {
            float hv = tanhf(P[(size_t)t * H_DIM + n] + acc[0]);
            states[(size_t)t * H_DIM + n] = hv;
            bf16_t hb = (bf16_t)hv;
            Sbf[(size_t)t * H_DIM + n]              = hb;
            hbuf[(size_t)((t + 1) & 1) * H_DIM + n] = hb;
        }
        __threadfence();
        __syncthreads();
        if (threadIdx.x == 0) {
            __hip_atomic_fetch_add(bar, 1u, __ATOMIC_RELEASE, __HIP_MEMORY_SCOPE_AGENT);
            const unsigned target = nb * (unsigned)(t + 1);
            while (__hip_atomic_load(bar, __ATOMIC_ACQUIRE, __HIP_MEMORY_SCOPE_AGENT) < target)
                __builtin_amdgcn_s_sleep(1);
        }
        __syncthreads();
        __threadfence();
    }
}

// ---------------------------------------------------------------------------
extern "C" void kernel_launch(void* const* d_in, const int* in_sizes, int n_in,
                              void* d_out, int out_size, void* d_ws, size_t ws_size,
                              hipStream_t stream) {
    (void)in_sizes; (void)n_in; (void)out_size; (void)ws_size;
    const float* inputs = (const float*)d_in[0];   // [T,1,I]
    const float* state  = (const float*)d_in[1];   // [1,H]
    const float* W_ih   = (const float*)d_in[2];   // [H,I]
    const float* W_hh   = (const float*)d_in[3];   // [H,H]
    const float* b_ih   = (const float*)d_in[4];   // [H]
    const float* W_out  = (const float*)d_in[5];   // [I, 1+I+H]

    float* outputs = (float*)d_out;                          // [T,I]
    float* states  = (float*)d_out + (size_t)T_DIM * I_DIM;  // [T,H]

    char* ws = (char*)d_ws;
    size_t off = 0;
    auto alloc = [&](size_t bytes) -> void* {
        void* p = ws + off;
        off = (off + bytes + 255) & ~(size_t)255;
        return p;
    };
    bf16_t*   Xbf   = (bf16_t*)alloc((size_t)T_DIM * I_DIM * 2);   //  8 MB
    bf16_t*   WihP  = (bf16_t*)alloc((size_t)H_DIM * I_DIM * 2);   //  8 MB
    bf16_t*   WhhP  = (bf16_t*)alloc((size_t)H_DIM * H_DIM * 2);   // 32 MB
    bf16_t*   WxP   = (bf16_t*)alloc((size_t)I_DIM * I_DIM * 2);   //  2 MB
    bf16_t*   WhP   = (bf16_t*)alloc((size_t)I_DIM * H_DIM * 2);   //  8 MB
    float*    Pm    = (float*) alloc((size_t)T_DIM * H_DIM * 4);   // 64 MB
    bf16_t*   Sbf   = (bf16_t*)alloc((size_t)T_DIM * H_DIM * 2);   // 32 MB
    float*    biasO = (float*) alloc((size_t)I_DIM * 4);
    bf16_t*   hbuf  = (bf16_t*)alloc((size_t)2 * H_DIM * 2);
    unsigned* bar   = (unsigned*)alloc(256);

    // 0) init h0 / bias / barrier
    esn_init<<<(H_DIM + 255) / 256, 256, 0, stream>>>(state, W_out, hbuf, biasO, bar);

    // 1) X -> bf16
    {
        long nel = (long)T_DIM * I_DIM;
        conv_bf16<<<(int)((nel + 255) / 256), 256, 0, stream>>>(inputs, Xbf, nel);
    }
    // 2) pack weights into WMMA B-fragment order (bf16)
    {
        long tot = (long)H_DIM * I_DIM;
        pack_b<<<(int)((tot + 255) / 256), 256, 0, stream>>>(W_ih, I_DIM, 0, WihP, I_DIM / 32, tot);
    }
    {
        long tot = (long)H_DIM * H_DIM;
        pack_b<<<(int)((tot + 255) / 256), 256, 0, stream>>>(W_hh, H_DIM, 0, WhhP, H_DIM / 32, tot);
    }
    {
        long tot = (long)I_DIM * I_DIM;
        pack_b<<<(int)((tot + 255) / 256), 256, 0, stream>>>(W_out, EXT_LD, 1, WxP, I_DIM / 32, tot);
    }
    {
        long tot = (long)I_DIM * H_DIM;
        pack_b<<<(int)((tot + 255) / 256), 256, 0, stream>>>(W_out, EXT_LD, 1 + I_DIM, WhP, H_DIM / 32, tot);
    }

    // 3) P = X * W_ih^T + b_ih          (M=T, N=H, K=I)
    {
        dim3 g(H_DIM / 128, T_DIM / 32);
        gemm_wmma_bf16<<<g, 256, 0, stream>>>(Xbf, I_DIM, WihP, Pm, H_DIM, I_DIM, b_ih, 0);
    }
    // 4) sequential recurrence -> states (fp32) and Sbf (bf16)
    esn_recurrence<<<32, 256, 0, stream>>>(WhhP, Pm, states, Sbf, hbuf, bar);

    // 5) outputs = bias + X * Wx^T      (M=T, N=I, K=I)
    {
        dim3 g(I_DIM / 128, T_DIM / 32);
        gemm_wmma_bf16<<<g, 256, 0, stream>>>(Xbf, I_DIM, WxP, outputs, I_DIM, I_DIM, biasO, 0);
    }
    // 6) outputs += S * Wh^T            (M=T, N=I, K=H)
    {
        dim3 g(I_DIM / 128, T_DIM / 32);
        gemm_wmma_bf16<<<g, 256, 0, stream>>>(Sbf, H_DIM, WhP, outputs, I_DIM, H_DIM, nullptr, 1);
    }
}
